// FusionBlock_31439160607466
// MI455X (gfx1250) — matrix operand order repaired and checked by reading, not compiled, not run on hardware
//
#include <hip/hip_runtime.h>
#include <hip/hip_bf16.h>

typedef __bf16 bf16;
typedef __attribute__((ext_vector_type(8)))  bf16  v8bf;
typedef __attribute__((ext_vector_type(16))) bf16  v16bf;
typedef __attribute__((ext_vector_type(8)))  float v8f;
typedef __attribute__((ext_vector_type(4)))  int   v4i;

// Concat two 8-element halves into a WMMA operand without per-element moves.
__device__ __forceinline__ v16bf concat16(v8bf lo, v8bf hi) {
    return __builtin_shufflevector(lo, hi, 0, 1, 2, 3, 4, 5, 6, 7,
                                   8, 9, 10, 11, 12, 13, 14, 15);
}

// CDNA5 async global->LDS copy path (ASYNCcnt-tracked), guarded so the file
// still compiles on toolchains that lack the builtins.
#if defined(__gfx1250__) && \
    __has_builtin(__builtin_amdgcn_global_load_async_to_lds_b128) && \
    __has_builtin(__builtin_amdgcn_s_wait_asynccnt)
#define USE_ASYNC_COPY 1
#define GAS1(p) ((__attribute__((address_space(1))) v4i*)(void*)(p))
#define LAS3(p) ((__attribute__((address_space(3))) v4i*)(void*)(p))
#else
#define USE_ASYNC_COPY 0
#endif

// ---------------------------------------------------------------------------
// f32 -> bf16 cast (weights)
// ---------------------------------------------------------------------------
__global__ __launch_bounds__(256) void cast_bf16_kernel(const float* __restrict__ in,
                                                        bf16* __restrict__ out, int n) {
    int i = blockIdx.x * blockDim.x + threadIdx.x;
    int stride = gridDim.x * blockDim.x;
    for (; i < n; i += stride) out[i] = (bf16)in[i];
}

// ---------------------------------------------------------------------------
// LayerNorm over rows of 1024 f32 -> bf16  (one block of 256 threads per row)
// ---------------------------------------------------------------------------
__global__ __launch_bounds__(256) void ln_kernel(const float* __restrict__ x,
                                                 const float* __restrict__ w,
                                                 const float* __restrict__ bvec,
                                                 bf16* __restrict__ out) {
    const int row = blockIdx.x;
    const int t = threadIdx.x;
    const float* xr = x + (size_t)row * 1024;
    float4 v = ((const float4*)xr)[t];
    float s  = v.x + v.y + v.z + v.w;
    float ss = v.x * v.x + v.y * v.y + v.z * v.z + v.w * v.w;
#pragma unroll
    for (int i = 1; i < 32; i <<= 1) {
        s  += __shfl_xor(s,  i, 32);
        ss += __shfl_xor(ss, i, 32);
    }
    __shared__ float sh[16];
    const int wave = t >> 5, lane = t & 31;
    if (lane == 0) { sh[wave] = s; sh[8 + wave] = ss; }
    __syncthreads();
    float st = 0.f, sst = 0.f;
#pragma unroll
    for (int i = 0; i < 8; ++i) { st += sh[i]; sst += sh[8 + i]; }
    const float mean = st * (1.f / 1024.f);
    const float var  = sst * (1.f / 1024.f) - mean * mean;
    const float rstd = rsqrtf(var + 1e-5f);
    float xv[4] = {v.x, v.y, v.z, v.w};
#pragma unroll
    for (int i = 0; i < 4; ++i) {
        int c = t * 4 + i;
        out[(size_t)row * 1024 + c] = (bf16)((xv[i] - mean) * rstd * w[c] + bvec[c]);
    }
}

// ---------------------------------------------------------------------------
// Tiled bf16 GEMM:  C[M,O] = act(A[M,K] * W[O,K]^T + bias) (+res)
// 256 threads = 8 waves; block tile 128x128, K-step 32; wave tile 32x64.
// Double-buffered LDS; async copies for tile t+1 overlap WMMAs of tile t.
// ---------------------------------------------------------------------------
template <bool GELU, bool HASB, bool HASR, bool OUTBF>
__global__ __launch_bounds__(256) void gemm_bf16_kernel(
    const bf16* __restrict__ A, const bf16* __restrict__ W,
    const float* __restrict__ bias, const float* __restrict__ res,
    float* __restrict__ outf, bf16* __restrict__ outb,
    int M, int K, int O) {
    constexpr int BK = 32, LSTR = 40;  // LDS row stride: 80B, 16B aligned
    __shared__ bf16 sA[2][128 * LSTR];
    __shared__ bf16 sW[2][128 * LSTR];
    const int t = threadIdx.x;
    const int wave = t >> 5, lane = t & 31;
    const int l16 = lane & 15, half = lane >> 4;
    const int wm = wave & 3, wo = wave >> 2;  // 4 (M) x 2 (O) wave grid
    const int bm = blockIdx.x * 128, bo = blockIdx.y * 128;
    const int lr = t >> 1, lc = (t & 1) * 16;  // cooperative tile load coords

    v8f acc[2][4] = {};

#if USE_ASYNC_COPY
    auto issue_copy = [&](int k0, int buf) {
        const bf16* ga = A + (size_t)(bm + lr) * K + k0 + lc;
        const bf16* gw = W + (size_t)(bo + lr) * K + k0 + lc;
        __builtin_amdgcn_global_load_async_to_lds_b128(
            GAS1(ga),     LAS3(&sA[buf][lr * LSTR + lc]),     0, 0);
        __builtin_amdgcn_global_load_async_to_lds_b128(
            GAS1(ga + 8), LAS3(&sA[buf][lr * LSTR + lc + 8]), 0, 0);
        __builtin_amdgcn_global_load_async_to_lds_b128(
            GAS1(gw),     LAS3(&sW[buf][lr * LSTR + lc]),     0, 0);
        __builtin_amdgcn_global_load_async_to_lds_b128(
            GAS1(gw + 8), LAS3(&sW[buf][lr * LSTR + lc + 8]), 0, 0);
    };
    issue_copy(0, 0);
#endif

    for (int k0 = 0; k0 < K; k0 += BK) {
        const int cur = (k0 >> 5) & 1;
#if USE_ASYNC_COPY
        __builtin_amdgcn_s_wait_asynccnt(0);  // tile `cur` resident in LDS
        __syncthreads();                      // all waves done reading buf cur^1
        if (k0 + BK < K) issue_copy(k0 + BK, cur ^ 1);  // overlap with WMMAs below
        if (k0 + 2 * BK < K) {                // warm L2 one tile further ahead
            __builtin_prefetch(A + (size_t)(bm + lr) * K + k0 + 2 * BK + lc, 0, 1);
            __builtin_prefetch(W + (size_t)(bo + lr) * K + k0 + 2 * BK + lc, 0, 1);
        }
#else
        const bf16* ga = A + (size_t)(bm + lr) * K + k0 + lc;
        const bf16* gw = W + (size_t)(bo + lr) * K + k0 + lc;
        v8bf a0 = *(const v8bf*)ga, a1 = *(const v8bf*)(ga + 8);
        v8bf w0 = *(const v8bf*)gw, w1 = *(const v8bf*)(gw + 8);
        __syncthreads();
        *(v8bf*)&sA[cur][lr * LSTR + lc] = a0;
        *(v8bf*)&sA[cur][lr * LSTR + lc + 8] = a1;
        *(v8bf*)&sW[cur][lr * LSTR + lc] = w0;
        *(v8bf*)&sW[cur][lr * LSTR + lc + 8] = w1;
        __syncthreads();
#endif

        // A fragments: lanes 0-15 hold K {0..7,16..23}; lanes 16-31 K {8..15,24..31}
        v16bf af[2];
#pragma unroll
        for (int mt = 0; mt < 2; ++mt) {
            const int row = wm * 32 + mt * 16 + l16;
            const int klo = half * 8;
            v8bf lo = *(const v8bf*)&sA[cur][row * LSTR + klo];
            v8bf hi = *(const v8bf*)&sA[cur][row * LSTR + klo + 16];
            af[mt] = concat16(lo, hi);
        }
#pragma unroll
        for (int ot = 0; ot < 4; ++ot) {
            // B fragment: lane n holds column n%16, K {0..7,16..23} (n<16) / {8..15,24..31}
            const int wrow = wo * 64 + ot * 16 + l16;
            const int kb = half * 8;
            v8bf lo = *(const v8bf*)&sW[cur][wrow * LSTR + kb];
            v8bf hi = *(const v8bf*)&sW[cur][wrow * LSTR + kb + 16];
            v16bf bfv = concat16(lo, hi);
#pragma unroll
            for (int mt = 0; mt < 2; ++mt)
                acc[mt][ot] = __builtin_amdgcn_wmma_f32_16x16x32_bf16(
                    false, af[mt], false, bfv, (short)0, acc[mt][ot], false, false);
        }
    }

    // epilogue: C layout -> lane l comp r: row = r + (l>>4)*8, col = l&15
#pragma unroll
    for (int mt = 0; mt < 2; ++mt)
#pragma unroll
        for (int ot = 0; ot < 4; ++ot) {
            const int col = bo + wo * 64 + ot * 16 + l16;
            float bv = HASB ? bias[col] : 0.f;
#pragma unroll
            for (int r = 0; r < 8; ++r) {
                const int row = bm + wm * 32 + mt * 16 + half * 8 + r;
                float v = acc[mt][ot][r] + bv;
                if (GELU) v = 0.5f * v * (1.0f + erff(v * 0.70710678118f));
                if (HASR) v += res[(size_t)row * O + col];
                if (OUTBF) outb[(size_t)row * O + col] = (bf16)v;
                else       outf[(size_t)row * O + col] = v;
            }
        }
}

// ---------------------------------------------------------------------------
// Fused flash attention.  qkv layout: [B,N,3,H,64] bf16 (row stride 3072).
// Block = 128 threads (4 waves); block covers (b,h, 64 q-rows); wave = 16 rows.
// Online softmax over key chunks of 32; S and P*V through WMMA.
// ---------------------------------------------------------------------------
__global__ __launch_bounds__(128) void attn_kernel(const bf16* __restrict__ qkv,
                                                   const int* __restrict__ amask,
                                                   bf16* __restrict__ outb) {
    const int bh = blockIdx.x;
    const int b = bh >> 4, h = bh & 15;
    const int qbase = blockIdx.y * 64;
    const int t = threadIdx.x, wave = t >> 5, lane = t & 31;
    const int l16 = lane & 15, half = lane >> 4;
    const size_t rowstr = 3072;

    __shared__ bf16 pT[4][16 * 32];  // per-wave P tile (C-layout -> A-layout relay)
    __shared__ bf16 vT[64 * 32];     // V chunk transposed: vT[ch][key]

    const bf16* qptr = qkv + (size_t)b * 1024 * rowstr + 0 * 1024 + h * 64;
    const bf16* kptr = qkv + (size_t)b * 1024 * rowstr + 1 * 1024 + h * 64;
    const bf16* vptr = qkv + (size_t)b * 1024 * rowstr + 2 * 1024 + h * 64;

    // Q fragments for this wave's 16 rows (two K-chunks of 32 channels)
    v16bf aq[2];
    {
        const int row = qbase + wave * 16 + l16;
#pragma unroll
        for (int kc = 0; kc < 2; ++kc) {
            const int c0 = kc * 32 + half * 8;
            const bf16* p = qptr + (size_t)row * rowstr + c0;
            aq[kc] = concat16(*(const v8bf*)p, *(const v8bf*)(p + 16));
        }
    }

    float mrow[8], lrow[8];
    v8f oacc[4] = {};
#pragma unroll
    for (int r = 0; r < 8; ++r) { mrow[r] = -1e30f; lrow[r] = 0.f; }
    const float scale = 0.125f;  // 64^-0.5
    const int mb = b * 1024;

    for (int kb = 0; kb < 1024; kb += 32) {
        // ---- S = Q K^T for two 16-key tiles ----
        v8f s[2] = {};
#pragma unroll
        for (int kt = 0; kt < 2; ++kt)
#pragma unroll
            for (int kc = 0; kc < 2; ++kc) {
                const int key = kb + kt * 16 + l16;
                const int c0 = kc * 32 + half * 8;
                const bf16* p = kptr + (size_t)key * rowstr + c0;
                v16bf bk = concat16(*(const v8bf*)p, *(const v8bf*)(p + 16));
                s[kt] = __builtin_amdgcn_wmma_f32_16x16x32_bf16(
                    false, aq[kc], false, bk, (short)0, s[kt], false, false);
            }
        const int m0 = amask[mb + kb + l16];
        const int m1 = amask[mb + kb + 16 + l16];

        __syncthreads();  // previous chunk's pT/vT reads complete
        // ---- stage V chunk transposed into LDS ----
        {
            const int key = t >> 2, cb = (t & 3) * 16;
            const bf16* p = vptr + (size_t)(kb + key) * rowstr + cb;
            v8bf v0 = *(const v8bf*)p, v1 = *(const v8bf*)(p + 8);
#pragma unroll
            for (int i = 0; i < 8; ++i) {
                vT[(cb + i) * 32 + key] = v0[i];
                vT[(cb + 8 + i) * 32 + key] = v1[i];
            }
        }
        // ---- online softmax (row stats across 16 lanes via shfl_xor) ----
#pragma unroll
        for (int r = 0; r < 8; ++r) {
            float s0 = s[0][r] * scale; if (!m0) s0 = -1e30f;
            float s1 = s[1][r] * scale; if (!m1) s1 = -1e30f;
            float tm = fmaxf(s0, s1);
#pragma unroll
            for (int i = 1; i < 16; i <<= 1) tm = fmaxf(tm, __shfl_xor(tm, i, 32));
            const float mn = fmaxf(mrow[r], tm);
            const float corr = __expf(mrow[r] - mn);
            const float p0 = __expf(s0 - mn), p1 = __expf(s1 - mn);
            float rs = p0 + p1;
#pragma unroll
            for (int i = 1; i < 16; i <<= 1) rs += __shfl_xor(rs, i, 32);
            lrow[r] = lrow[r] * corr + rs;
            mrow[r] = mn;
#pragma unroll
            for (int ct = 0; ct < 4; ++ct) oacc[ct][r] *= corr;
            pT[wave][(half * 8 + r) * 32 + l16] = (bf16)p0;
            pT[wave][(half * 8 + r) * 32 + 16 + l16] = (bf16)p1;
        }
        __syncthreads();

        // ---- O += P V ----
        v16bf ap;
        {
            const int klo = half * 8;
            v8bf lo = *(const v8bf*)&pT[wave][l16 * 32 + klo];
            v8bf hi = *(const v8bf*)&pT[wave][l16 * 32 + klo + 16];
            ap = concat16(lo, hi);
        }
#pragma unroll
        for (int ct = 0; ct < 4; ++ct) {
            const int ch = ct * 16 + l16;
            v8bf lo = *(const v8bf*)&vT[ch * 32 + half * 8];
            v8bf hi = *(const v8bf*)&vT[ch * 32 + half * 8 + 16];
            v16bf bv = concat16(lo, hi);
            oacc[ct] = __builtin_amdgcn_wmma_f32_16x16x32_bf16(
                false, ap, false, bv, (short)0, oacc[ct], false, false);
        }
    }

    // ---- normalize and write [B*N, 1024] bf16 at col h*64+ch ----
#pragma unroll
    for (int r = 0; r < 8; ++r) {
        const float inv = 1.0f / lrow[r];
        const int row = qbase + wave * 16 + half * 8 + r;
#pragma unroll
        for (int ct = 0; ct < 4; ++ct) {
            const int ch = h * 64 + ct * 16 + l16;
            outb[(size_t)(b * 1024 + row) * 1024 + ch] = (bf16)(oacc[ct][r] * inv);
        }
    }
}

// ---------------------------------------------------------------------------
// Host launcher
// ---------------------------------------------------------------------------
extern "C" void kernel_launch(void* const* d_in, const int* in_sizes, int n_in,
                              void* d_out, int out_size, void* d_ws, size_t ws_size,
                              hipStream_t stream) {
    (void)in_sizes; (void)n_in; (void)out_size; (void)ws_size;
    const float* x       = (const float*)d_in[0];
    const int*   amask   = (const int*)d_in[1];
    const float* qkv_w   = (const float*)d_in[2];
    const float* proj_w  = (const float*)d_in[3];
    const float* proj_b  = (const float*)d_in[4];
    const float* fc1_w   = (const float*)d_in[5];
    const float* fc1_b   = (const float*)d_in[6];
    const float* fc2_w   = (const float*)d_in[7];
    const float* fc2_b   = (const float*)d_in[8];
    const float* norm1_w = (const float*)d_in[9];
    const float* norm1_b = (const float*)d_in[10];
    const float* norm2_w = (const float*)d_in[11];
    const float* norm2_b = (const float*)d_in[12];
    float* out = (float*)d_out;

    const int M = 4096;  // B*N
    char* ws = (char*)d_ws;
    size_t off = 0;
    auto take = [&](size_t bytes) { char* p = ws + off; off = (off + bytes + 255) & ~(size_t)255; return p; };
    bf16* hln    = (bf16*)take((size_t)M * 1024 * 2);   // LN output (reused for LN2)
    bf16* qkvbuf = (bf16*)take((size_t)M * 3072 * 2);
    bf16* attnb  = (bf16*)take((size_t)M * 1024 * 2);
    bf16* h1     = (bf16*)take((size_t)M * 4096 * 2);
    float* x1    = (float*)take((size_t)M * 1024 * 4);
    bf16* wq = (bf16*)take((size_t)3072 * 1024 * 2);
    bf16* wp = (bf16*)take((size_t)1024 * 1024 * 2);
    bf16* w1 = (bf16*)take((size_t)4096 * 1024 * 2);
    bf16* w2 = (bf16*)take((size_t)1024 * 4096 * 2);

    // 0) cast weights once per call (cheap vs compute)
    cast_bf16_kernel<<<2048, 256, 0, stream>>>(qkv_w, wq, 3072 * 1024);
    cast_bf16_kernel<<<2048, 256, 0, stream>>>(proj_w, wp, 1024 * 1024);
    cast_bf16_kernel<<<2048, 256, 0, stream>>>(fc1_w, w1, 4096 * 1024);
    cast_bf16_kernel<<<2048, 256, 0, stream>>>(fc2_w, w2, 1024 * 4096);

    // 1) LN1
    ln_kernel<<<M, 256, 0, stream>>>(x, norm1_w, norm1_b, hln);
    // 2) QKV GEMM -> bf16 [M, 3*H*64]
    gemm_bf16_kernel<false, false, false, true><<<dim3(32, 24), 256, 0, stream>>>(
        hln, wq, nullptr, nullptr, nullptr, qkvbuf, M, 1024, 3072);
    // 3) fused attention
    attn_kernel<<<dim3(64, 16), 128, 0, stream>>>(qkvbuf, amask, attnb);
    // 4) proj + bias + residual(x) -> x1 (f32)
    gemm_bf16_kernel<false, true, true, false><<<dim3(32, 8), 256, 0, stream>>>(
        attnb, wp, proj_b, x, x1, nullptr, M, 1024, 1024);
    // 5) LN2
    ln_kernel<<<M, 256, 0, stream>>>(x1, norm2_w, norm2_b, hln);
    // 6) FC1 + bias + exact GELU -> bf16
    gemm_bf16_kernel<true, true, false, true><<<dim3(32, 32), 256, 0, stream>>>(
        hln, w1, fc1_b, nullptr, nullptr, h1, M, 1024, 4096);
    // 7) FC2 + bias + residual(x1) -> d_out (f32)
    gemm_bf16_kernel<false, true, true, false><<<dim3(32, 8), 256, 0, stream>>>(
        h1, w2, fc2_b, x1, out, nullptr, M, 4096, 1024);
}
